// UHGMultiHeadAttention_73744588472874
// MI455X (gfx1250) — matrix operand order repaired and checked by reading, not compiled, not run on hardware
//
#include <hip/hip_runtime.h>

// ---------------------------------------------------------------------------
// UHG multi-head attention for MI455X (gfx1250), wave32 + WMMA f16->f32
// + Tensor Data Mover (TDM) double-buffered V staging in the fused attention.
// B=4, S=2048, D=256, H=4.
// ---------------------------------------------------------------------------

#define EPSF 1e-6f

typedef __attribute__((ext_vector_type(16))) _Float16 v16h;
typedef __attribute__((ext_vector_type(8)))  _Float16 v8h;
typedef __attribute__((ext_vector_type(8)))  float    v8f;
typedef unsigned int u32x4 __attribute__((ext_vector_type(4)));
typedef int          i32x8 __attribute__((ext_vector_type(8)));
typedef int          i32x4 __attribute__((ext_vector_type(4)));

union HV16 { v16h v; v8h h[2]; };

constexpr int B = 4, S = 2048, D = 256, H = 4;
constexpr int NMAT = 13;            // 4 Wq + 4 Wk + 4 Wv + 1 Wo (spatial blocks)
constexpr int MAT  = D * D;         // 65536
constexpr int VPAD = 272;           // 257 padded to multiple of 16
constexpr int KC   = 32;            // K-chunk per WMMA step

// ---------------------------------------------------------------------------
// TDM issue: 2D tile (tile_dim0=32 halves contiguous, tile_dim1=272 rows,
// row stride = S elements) from global vT slice into contiguous LDS.
// D# layout per CDNA5 ISA 8.3/8.4 (group0: count/lds/global/type,
// group1: data_size + dims/strides). Groups 2/3 unused (2D tensor).
// Toolchain here is the 6-arg clang-23 form (g0,g1,g2,g3,g4,cpol).
// ---------------------------------------------------------------------------
__device__ __forceinline__ void tdm_load_vtile(unsigned lds_addr,
                                               unsigned long long gaddr)
{
  u32x4 g0 = {
    1u,                                                  // count=1, user mode
    lds_addr,                                            // D#.lds_addr
    (unsigned)(gaddr & 0xFFFFFFFFull),                   // global_addr[31:0]
    (unsigned)((gaddr >> 32) & 0x01FFFFFFull) | (2u << 30) // addr[56:32]|type=2
  };
  i32x8 g1 = {
    0x00010000,                  // wg_mask=0, data_size=1 (2 bytes)
    (int)(((unsigned)S) << 16),  // tensor_dim0[15:0]=2048 at bits 63:48
    (int)(((unsigned)VPAD) << 16), // dim0[31:16]=0, tensor_dim1[15:0]=272
    (int)(((unsigned)KC) << 16), // dim1[31:16]=0, tile_dim0=32 at bits 127:112
    VPAD,                        // tile_dim1=272 (bits 143:128), tile_dim2=0
    S,                           // tensor_dim0_stride[31:0]=2048 (bits 191:160)
    0,                           // dim0_stride[47:32]=0, dim1_stride[15:0]=0
    0                            // tensor_dim1_stride rest = 0
  };
  i32x4 z4 = {0, 0, 0, 0};
  i32x8 z8 = {0, 0, 0, 0, 0, 0, 0, 0};
  __builtin_amdgcn_tensor_load_to_lds(g0, g1, z4, z4, z8, 0);
}

// ---------------------------------------------------------------------------
// Polar decomposition prep (Newton-Schulz). slogdet scaling provably cancels.
// ---------------------------------------------------------------------------
__global__ __launch_bounds__(256) void polar_init_kernel(
    const float* __restrict__ Wq, const float* __restrict__ Wk,
    const float* __restrict__ Wv, const float* __restrict__ Wo,
    float* __restrict__ X)
{
  int gid = blockIdx.x * 256 + threadIdx.x;     // NMAT * 65536 total
  int mat = gid >> 16;
  int rc  = gid & 65535;
  int r = rc >> 8, c = rc & 255;
  const float* src;
  if (mat < 4)       src = Wq + (size_t)mat * 257 * 257;
  else if (mat < 8)  src = Wk + (size_t)(mat - 4) * 257 * 257;
  else if (mat < 12) src = Wv + (size_t)(mat - 8) * 257 * 257;
  else               src = Wo;
  X[gid] = src[(size_t)r * 257 + c];
}

__global__ __launch_bounds__(256) void ns_ata_kernel(
    const float* __restrict__ X, float* __restrict__ Y)
{
  int gid = blockIdx.x * 256 + threadIdx.x;
  int mat = gid >> 16;
  int rc  = gid & 65535;
  int r = rc >> 8, c = rc & 255;
  const float* Xm = X + (size_t)mat * MAT;
  float acc = 0.0f;
  for (int k = 0; k < D; ++k)
    acc = fmaf(Xm[k * D + r], Xm[k * D + c], acc);   // (X^T X)[r][c]
  Y[gid] = acc;
}

__global__ __launch_bounds__(256) void ns_update_kernel(
    const float* __restrict__ X, const float* __restrict__ Y,
    float* __restrict__ Xn)
{
  int gid = blockIdx.x * 256 + threadIdx.x;
  int mat = gid >> 16;
  int rc  = gid & 65535;
  int r = rc >> 8, c = rc & 255;
  const float* Xm = X + (size_t)mat * MAT;
  const float* Ym = Y + (size_t)mat * MAT;
  float acc = 0.0f;
  for (int k = 0; k < D; ++k)
    acc = fmaf(Xm[r * D + k], Ym[k * D + c], acc);   // (X * X^T X)[r][c]
  Xn[gid] = 1.5f * Xm[r * D + c] - 0.5f * acc;
}

// fp32 -> f16 convert (used for weights and for the value tensor)
__global__ __launch_bounds__(256) void cvt_f16_kernel(
    const float* __restrict__ src, _Float16* __restrict__ dst, int n)
{
  int gid = blockIdx.x * 256 + threadIdx.x;
  if (gid < n) dst[gid] = (_Float16)src[gid];
}

// ---------------------------------------------------------------------------
// q/k path: only columns {0,1,2} of the transform matter (t_last == 1).
// ---------------------------------------------------------------------------
__global__ __launch_bounds__(256) void qk_kernel(
    const float* __restrict__ X,   // [B*S, 256]
    const float* __restrict__ ws,  // 4 heads, each [256,256] row-major
    float* __restrict__ out)       // [B*H, S, 2]
{
  int gid = blockIdx.x * 256 + threadIdx.x;   // H * B*S
  int h   = gid >> 13;
  int row = gid & 8191;
  int b = row >> 11, s = row & 2047;
  const float* x = X + (size_t)row * D;
  const float* w = ws + (size_t)h * MAT;
  float t0 = 0.0f, t1 = 0.0f, t2 = 0.0f;
  for (int k = 0; k < D; ++k) {
    float xv = x[k];
    t0 = fmaf(xv, w[0 * D + k], t0);
    t1 = fmaf(xv, w[1 * D + k], t1);
    t2 = fmaf(xv, w[2 * D + k], t2);
  }
  const float inv = 1.0f / (1.0f + EPSF);
  float u0 = t0 * inv, u1 = t1 * inv, u2 = t2 * inv, ul = inv;
  float c1 = (ul - u1) / ((ul - u2) + EPSF);
  float d1 = ul + EPSF * (c1 + EPSF);
  float q0 = u0 / d1, q1 = u1 / d1, q2 = u2 / d1, ql = ul / d1;
  float c2 = (ql - q1) / ((ql - q2) + EPSF);
  float d2 = ql + EPSF * (c2 + EPSF);
  float* o = out + ((size_t)(b * H + h) * S + s) * 2;
  o[0] = q0 / d2;
  o[1] = q1 / d2;
}

// ---------------------------------------------------------------------------
// WMMA GEMM: C[head][M,256] = A[M,256](f16) * W[head](f16)^T, f32 accumulate.
// ---------------------------------------------------------------------------
__global__ __launch_bounds__(128) void proj_gemm_kernel(
    const _Float16* __restrict__ A,   // [M,256] row-major
    const _Float16* __restrict__ W,   // per-head [256,256] row-major
    size_t wstride, float* __restrict__ C, size_t cstride, int M)
{
  const int head = blockIdx.y;
  const int mt   = blockIdx.x;
  const int wave = threadIdx.x >> 5;
  const int lane = threadIdx.x & 31;
  const int lm   = lane & 15;
  const bool hi  = lane >= 16;
  const _Float16* Wm = W + (size_t)head * wstride;
  float* Cm = C + (size_t)head * cstride;

  v8f acc[4] = {};
  const int row = mt * 16 + lm;
  for (int k0 = 0; k0 < D; k0 += 32) {
    HV16 af;   // A frag: lane lm = row, K contiguous k0+16*hi .. +15
    const _Float16* ap = A + (size_t)row * D + k0 + (hi ? 16 : 0);
    af.h[0] = *(const v8h*)(ap);
    af.h[1] = *(const v8h*)(ap + 8);
#pragma unroll
    for (int i = 0; i < 4; ++i) {
      const int nt = wave * 4 + i;
      const _Float16* bp = Wm + (size_t)(nt * 16 + lm) * D + k0 + (hi ? 16 : 0);
      v16h bf = *(const v16h*)bp;
      acc[i] = __builtin_amdgcn_wmma_f32_16x16x32_f16(
          false, af.v, false, bf, (short)0, acc[i], false, false);
    }
  }
#pragma unroll
  for (int i = 0; i < 4; ++i) {
    const int col = (wave * 4 + i) * 16 + lm;
#pragma unroll
    for (int g = 0; g < 8; ++g) {
      int m = g + (hi ? 8 : 0);
      Cm[(size_t)(mt * 16 + m) * D + col] = acc[i][g];
    }
  }
}

// ---------------------------------------------------------------------------
// v epilogue: per-row cr-norm chain, write TRANSPOSED f16 value matrix
// vT[b,h, n, s] (n padded to 272, pad = 0).
// ---------------------------------------------------------------------------
__global__ __launch_bounds__(256) void v_finalize_kernel(
    const float* __restrict__ T,   // [H, B*S, 256]
    _Float16* __restrict__ vT)     // [B*H, VPAD, S]
{
  int gid = blockIdx.x * 256 + threadIdx.x;   // H * B*S
  int h   = gid >> 13;
  int row = gid & 8191;
  int b = row >> 11, s = row & 2047;
  const float* t = T + ((size_t)h * (B * S) + row) * D;
  const float inv = 1.0f / (1.0f + EPSF);
  float u1 = t[1] * inv, u2 = t[2] * inv, ul = inv;   // t_last == 1
  float c1 = (ul - u1) / ((ul - u2) + EPSF);
  float d1 = ul + EPSF * (c1 + EPSF);
  float sc = inv / d1;
  _Float16* dst = vT + (size_t)(b * H + h) * VPAD * S + s;
  for (int n = 0; n < D; ++n)
    dst[(size_t)n * S] = (_Float16)(t[n] * sc);
  dst[(size_t)D * S] = (_Float16)(ul / d1);           // homogeneous coord
  for (int n = D + 1; n < VPAD; ++n)
    dst[(size_t)n * S] = (_Float16)0.0f;              // pad
}

// ---------------------------------------------------------------------------
// Fused attention: scores -> exp -> softmax -> attn store -> P@V (WMMA with
// TDM double-buffered V tiles in LDS) -> L2 row-normalize -> f16 output.
// One block = 16 query rows of one (b,h). LDS ~101 KB (P 64K + 2 V tiles).
// ---------------------------------------------------------------------------
__global__ __launch_bounds__(256) void attn_kernel(
    const float* __restrict__ qq,     // [B*H, S, 2]
    const float* __restrict__ kk,     // [B*H, S, 2]
    const _Float16* __restrict__ vT,  // [B*H, VPAD, S]
    float* __restrict__ attn,         // [B*H, S, S]
    _Float16* __restrict__ wsn,       // [B, S, H, 256]
    float* __restrict__ plast)        // [B, S, H]
{
  __shared__ _Float16 p[16][S];          // 64 KB: exp(scores), unnormalized
  __shared__ _Float16 vtile[2][VPAD * KC]; // 2 x 17 KB TDM staging buffers
  __shared__ float qrow[16][2];
  __shared__ float redsum[16][8];
  __shared__ float rowsum[16];
  __shared__ float rowsq[16];

  const int bh  = blockIdx.y;
  const int b   = bh / H, h = bh % H;
  const int mt  = blockIdx.x;            // 16-row tile index
  const int tid = threadIdx.x;
  const int wave = tid >> 5, lane = tid & 31;

  if (tid < 16) {
    const float* qp = qq + ((size_t)bh * S + mt * 16 + tid) * 2;
    qrow[tid][0] = qp[0];
    qrow[tid][1] = qp[1];
    rowsq[tid] = 0.0f;
  }
  __syncthreads();

  // Phase A: scores + exp (scores in (0,1] so no max-shift needed)
  float part[16];
#pragma unroll
  for (int r = 0; r < 16; ++r) part[r] = 0.0f;
  for (int j = tid; j < S; j += 256) {
    const float* kp = kk + ((size_t)bh * S + j) * 2;
    float k0 = kp[0], k1 = kp[1];
#pragma unroll
    for (int r = 0; r < 16; ++r) {
      float cr = (qrow[r][0] * k1) / (qrow[r][1] * k0 + EPSF);
      float sc = 1.0f / (1.0f + fabsf(__logf(fabsf(cr) + EPSF)));
      float pe = __expf(sc);
      p[r][j] = (_Float16)pe;
      part[r] += pe;
    }
  }
#pragma unroll
  for (int r = 0; r < 16; ++r) {      // wave32 reduction, then cross-wave
    float v = part[r];
    for (int off = 16; off > 0; off >>= 1) v += __shfl_xor(v, off, 32);
    if (lane == 0) redsum[r][wave] = v;
  }
  __syncthreads();
  if (tid < 16) {
    float s = 0.0f;
#pragma unroll
    for (int w = 0; w < 8; ++w) s += redsum[tid][w];
    rowsum[tid] = s;
  }
  __syncthreads();

  // Phase B: write normalized attention (mandatory 268 MB output)
  for (int idx = tid; idx < 16 * S; idx += 256) {
    int r = idx >> 11, j = idx & (S - 1);
    attn[((size_t)bh * S + mt * 16 + r) * S + j] = (float)p[r][j] / rowsum[r];
  }

  // Phase C: ws = P @ V. K-outer loop; V K-chunks DMA'd by the TDM into LDS
  // (double-buffered, TENSORcnt-tracked); 8 waves cover 17 N-tiles.
  const int lm = lane & 15;
  const bool hiL = lane >= 16;
  v8f acc[3] = {};
  int nts[3] = {0, 0, 0};
  int ncnt = 0;
  for (int nt = wave; nt < 17; nt += 8) nts[ncnt++] = nt;

  const unsigned lds0 = (unsigned)(unsigned long long)(void*)&vtile[0][0];
  const unsigned lds1 = (unsigned)(unsigned long long)(void*)&vtile[1][0];
  const unsigned long long gbase =
      (unsigned long long)(const void*)(vT + (size_t)bh * VPAD * S);

  if (wave == 0) tdm_load_vtile(lds0, gbase);          // prime buffer 0
  for (int it = 0; it < S / KC; ++it) {
    if (wave == 0) __builtin_amdgcn_s_wait_tensorcnt(0); // current tile landed
    __syncthreads();                                    // publish to all waves
    if (wave == 0 && it + 1 < S / KC)                   // prefetch next chunk
      tdm_load_vtile(((it + 1) & 1) ? lds1 : lds0,
                     gbase + (unsigned long long)(it + 1) * KC * 2);

    const _Float16* vt = (it & 1) ? &vtile[1][0] : &vtile[0][0];
    HV16 af;
    const _Float16* ap = &p[lm][it * KC + (hiL ? 16 : 0)];
    af.h[0] = *(const v8h*)ap;
    af.h[1] = *(const v8h*)(ap + 8);
#pragma unroll
    for (int c = 0; c < 3; ++c) {
      if (c < ncnt) {
        const _Float16* bp = vt + (nts[c] * 16 + lm) * KC + (hiL ? 16 : 0);
        v16h bf = *(const v16h*)bp;
        acc[c] = __builtin_amdgcn_wmma_f32_16x16x32_f16(
            false, af.v, false, bf, (short)0, acc[c], false, false);
      }
    }
  }

  for (int c = 0; c < ncnt; ++c) {      // row-square partials for L2 norm
#pragma unroll
    for (int g = 0; g < 8; ++g) {
      int m = g + (hiL ? 8 : 0);
      atomicAdd(&rowsq[m], acc[c][g] * acc[c][g]);
    }
  }
  __syncthreads();

  // Epilogue: out = acc / (||acc|| + eps*rowsum)  (softmax 1/rowsum folded in)
  for (int c = 0; c < ncnt; ++c) {
    const int col = nts[c] * 16 + lm;
#pragma unroll
    for (int g = 0; g < 8; ++g) {
      int m = g + (hiL ? 8 : 0);
      float scale = 1.0f / (sqrtf(rowsq[m]) + EPSF * rowsum[m]);
      float val = acc[c][g] * scale;
      int srow = mt * 16 + m;
      if (col < 256)
        wsn[(((size_t)b * S + srow) * H + h) * 256 + col] = (_Float16)val;
      else if (col == 256)
        plast[((size_t)b * S + srow) * H + h] = val;
    }
  }
}

// ---------------------------------------------------------------------------
// Final epilogue after the W_o GEMM: per-row cr-norm chain, write [..,:256].
// ---------------------------------------------------------------------------
__global__ __launch_bounds__(256) void out_finalize_kernel(
    const float* __restrict__ T,      // [B*S*H, 256]
    const float* __restrict__ plast,  // [B*S*H]
    float* __restrict__ out)          // [B, S*H, 256]
{
  __shared__ float sc;
  const int row = blockIdx.x;
  const float* t = T + (size_t)row * D;
  if (threadIdx.x == 0) {
    float tl = plast[row];
    float i1 = 1.0f / (tl + EPSF);
    float u1 = t[1] * i1, u2 = t[2] * i1, ul = tl * i1;
    float c1 = (ul - u1) / ((ul - u2) + EPSF);
    sc = i1 / (ul + EPSF * (c1 + EPSF));
  }
  __syncthreads();
  out[(size_t)row * D + threadIdx.x] = t[threadIdx.x] * sc;
}

// ---------------------------------------------------------------------------
extern "C" void kernel_launch(void* const* d_in, const int* in_sizes, int n_in,
                              void* d_out, int out_size, void* d_ws,
                              size_t ws_size, hipStream_t stream)
{
  (void)in_sizes; (void)n_in; (void)out_size; (void)ws_size;
  const float* query = (const float*)d_in[0];
  const float* key   = (const float*)d_in[1];
  const float* value = (const float*)d_in[2];
  const float* Wq    = (const float*)d_in[3];
  const float* Wk    = (const float*)d_in[4];
  const float* Wv    = (const float*)d_in[5];
  const float* Wo    = (const float*)d_in[6];

  // scratch carve-up (all chunks 256B-aligned by construction; ~81 MiB)
  float*    XA    = (float*)d_ws;
  float*    XB    = XA + (size_t)NMAT * MAT;
  float*    Y     = XB + (size_t)NMAT * MAT;
  _Float16* WH    = (_Float16*)(Y + (size_t)NMAT * MAT);
  _Float16* VAL16 = WH + (size_t)NMAT * MAT;
  float*    QQ    = (float*)(VAL16 + (size_t)B * S * D);
  float*    KK    = QQ + (size_t)B * H * S * 2;
  float*    TV    = KK + (size_t)B * H * S * 2;       // reused for t_o later
  _Float16* VT    = (_Float16*)(TV + (size_t)H * B * S * D);
  _Float16* WSN   = VT + (size_t)B * H * VPAD * S;
  float*    PLAST = (float*)(WSN + (size_t)B * S * H * D);

  const int nsBlocks = (NMAT * MAT) / 256;            // 3328

  // 1) polar factors via Newton-Schulz (slogdet scaling cancels exactly)
  polar_init_kernel<<<nsBlocks, 256, 0, stream>>>(Wq, Wk, Wv, Wo, XA);
  float* cur = XA; float* nxt = XB;
  for (int i = 0; i < 8; ++i) {
    ns_ata_kernel<<<nsBlocks, 256, 0, stream>>>(cur, Y);
    ns_update_kernel<<<nsBlocks, 256, 0, stream>>>(cur, Y, nxt);
    float* tmp = cur; cur = nxt; nxt = tmp;
  }

  // 2) f16 packs
  cvt_f16_kernel<<<nsBlocks, 256, 0, stream>>>(cur, WH, NMAT * MAT);
  cvt_f16_kernel<<<(B * S * D) / 256, 256, 0, stream>>>(value, VAL16, B * S * D);

  // 3) q/k reduced transforms (3 columns suffice)
  qk_kernel<<<(H * B * S) / 256, 256, 0, stream>>>(query, cur, QQ);
  qk_kernel<<<(H * B * S) / 256, 256, 0, stream>>>(key, cur + (size_t)4 * MAT, KK);

  // 4) v transform (WMMA GEMM per head) + transpose/pad epilogue
  proj_gemm_kernel<<<dim3((B * S) / 16, H), 128, 0, stream>>>(
      VAL16, WH + (size_t)8 * MAT, MAT, TV, (size_t)B * S * D, B * S);
  v_finalize_kernel<<<(H * B * S) / 256, 256, 0, stream>>>(TV, VT);

  // 5) fused attention (softmax + attn output + TDM-staged P@V + L2 norm)
  float* attn_out = (float*)d_out + (size_t)B * S * H * D;
  attn_kernel<<<dim3(S / 16, B * H), 256, 0, stream>>>(
      QQ, KK, VT, attn_out, WSN, PLAST);

  // 6) W_o transform (WMMA GEMM) + final cr-norm epilogue
  proj_gemm_kernel<<<dim3((B * S * H) / 16, 1), 128, 0, stream>>>(
      WSN, WH + (size_t)12 * MAT, 0, TV, 0, B * S * H);
  out_finalize_kernel<<<B * S * H, 256, 0, stream>>>(TV, PLAST, (float*)d_out);
}
